// SARDecoder_41953240547657
// MI455X (gfx1250) — matrix operand order repaired and checked by reading, not compiled, not run on hardware
//
#include <hip/hip_runtime.h>
#include <hip/hip_bf16.h>
#include <math.h>

// ---------------------------------------------------------------------------
// Types for CDNA5 WMMA (wave32): v_wmma_f32_16x16x32_bf16
// ---------------------------------------------------------------------------
typedef __bf16 bf16;
typedef __attribute__((ext_vector_type(8)))  bf16  bf16x8;
typedef __attribute__((ext_vector_type(16))) bf16  bf16x16;
typedef __attribute__((ext_vector_type(8)))  float f32x8;

#define B_   16
#define T_   30
#define BT_  (B_ * T_)      // 480
#define FEAT_ 512
#define RNN_  512
#define ATTN_ 512
#define G_    2048          // 4*RNN
#define HW_   512           // 8*64
#define VOC_  111           // vocab+1
#define KC_   4608          // 512*9 conv K
#define NB   4              // N-blocking for big GEMMs / conv

__device__ __forceinline__ float sigm(float x) { return 1.0f / (1.0f + __expf(-x)); }
// tanh via one hardware v_exp_f32 (TRANS pipe); exact at both tails.
__device__ __forceinline__ float ftanh(float x) {
    float e = __expf(2.0f * x);
    return 1.0f - 2.0f / (e + 1.0f);
}

// A fragment: 16x32 bf16, A row-major [M,K].
// lane l, elem e: m = l&15, k = (e>>3)*16 + (l>=16?8:0) + (e&7)
__device__ __forceinline__ bf16x16 load_a_frag(const bf16* __restrict__ A, int K,
                                               int row, int kt, int lane) {
    const bf16* p = A + (size_t)row * K + kt * 32 + ((lane & 16) ? 8 : 0);
    bf16x8 lo = *(const bf16x8*)p;
    bf16x8 hi = *(const bf16x8*)(p + 16);
    return __builtin_shufflevector(lo, hi, 0,1,2,3,4,5,6,7,8,9,10,11,12,13,14,15);
}

// B fragment: 32x16 bf16 from W row-major [N,K] (B[k][n] = W[n][k]).
// lane l, elem e: n = l&15, k = (e>>3)*16 + (l>=16?8:0) + (e&7)
__device__ __forceinline__ bf16x16 load_b_frag(const bf16* __restrict__ W, int K,
                                               int nrow, int kt, int lane) {
    const bf16* p = W + (size_t)nrow * K + kt * 32 + ((lane & 16) ? 8 : 0);
    bf16x8 lo = *(const bf16x8*)p;
    bf16x8 hi = *(const bf16x8*)(p + 16);
    return __builtin_shufflevector(lo, hi, 0,1,2,3,4,5,6,7,8,9,10,11,12,13,14,15);
}

// ---------------------------------------------------------------------------
// Generic WMMA GEMM, compile-time 1xNBT register-blocked along N.
// REQUIRES (N/16) % NBT == 0 — no guards in the k-loop, so no exec masking
// around WMMA and no accumulator spilling. C = A[M,K] @ W[N,K]^T (+bias)
// ---------------------------------------------------------------------------
template <int NBT>
__global__ void k_gemm_t(const bf16* __restrict__ A, const bf16* __restrict__ W,
                         float* __restrict__ C, const float* __restrict__ bias,
                         int M, int N, int K, int ldC, int Nstore) {
    int wave = (int)((blockIdx.x * blockDim.x + threadIdx.x) >> 5);
    int lane = (int)(threadIdx.x & 31);
    int Mt = M >> 4, Nt = N >> 4;
    int Ng = Nt / NBT;
    if (wave >= Mt * Ng) return;       // wave-uniform
    int mt = wave / Ng, ng = wave % Ng;

    f32x8 acc[NBT] = {};
    int arow = mt * 16 + (lane & 15);
    int ncol = lane & 15;
    int ksteps = K >> 5;
    for (int kt = 0; kt < ksteps; ++kt) {
        bf16x16 a = load_a_frag(A, K, arow, kt, lane);
#pragma unroll
        for (int j = 0; j < NBT; ++j) {
            bf16x16 b = load_b_frag(W, K, (ng * NBT + j) * 16 + ncol, kt, lane);
            acc[j] = __builtin_amdgcn_wmma_f32_16x16x32_bf16(
                false, a, false, b, (short)0, acc[j], false, false);
        }
    }
    int mbase = mt * 16 + ((lane & 16) ? 8 : 0);
#pragma unroll
    for (int j = 0; j < NBT; ++j) {
        int n = (ng * NBT + j) * 16 + ncol;
        if (n < Nstore) {              // store-side divergence only (after WMMA)
            float bv = bias ? bias[n] : 0.0f;
#pragma unroll
            for (int v = 0; v < 8; ++v)
                C[(size_t)(mbase + v) * ldC + n] = acc[j][v] + bv;
        }
    }
}

// ---------------------------------------------------------------------------
// Implicit-GEMM 3x3 same-pad conv, 1x4 N-blocked: P = im2col(Fbhwc) @ Wc^T + b
// Fbhwc: [16,8,64,512] bf16 ; Wc: [512 outc, 4608] bf16 with k = tap*512 + ic
// ---------------------------------------------------------------------------
__global__ void k_conv(const bf16* __restrict__ Fb, const bf16* __restrict__ Wc,
                       const float* __restrict__ bias, float* __restrict__ P) {
    int wave = (int)((blockIdx.x * blockDim.x + threadIdx.x) >> 5);
    int lane = (int)(threadIdx.x & 31);
    if (wave >= 512 * 8) return;        // Mt=512, Ngroups=32/4=8
    int mt = wave >> 3, ng = wave & 7;

    int m = mt * 16 + (lane & 15);      // row = b*512 + y*64 + x
    int b = m >> 9, y = (m >> 6) & 7, x = m & 63;
    int ncol = lane & 15;

    f32x8 acc[NB] = {};
    for (int kt = 0; kt < 144; ++kt) {  // K = 4608 = 144*32
        int tap = kt >> 4;              // 16 k-steps per tap (512/32)
        int kh = tap / 3, kw = tap % 3;
        int iy = y + kh - 1, ix = x + kw - 1;
        bf16x16 a = {};
        if ((unsigned)iy < 8u && (unsigned)ix < 64u) {  // lane-divergent loads only;
            const bf16* p = Fb + ((((size_t)b * 8 + iy) * 64 + ix) << 9)
                               + ((kt & 15) << 5) + ((lane & 16) ? 8 : 0);
            bf16x8 lo = *(const bf16x8*)p;
            bf16x8 hi = *(const bf16x8*)(p + 16);
            a = __builtin_shufflevector(lo, hi, 0,1,2,3,4,5,6,7,8,9,10,11,12,13,14,15);
        }                               // reconverged before the WMMAs below
#pragma unroll
        for (int j = 0; j < NB; ++j) {
            bf16x16 bb = load_b_frag(Wc, KC_, (ng * NB + j) * 16 + ncol, kt, lane);
            acc[j] = __builtin_amdgcn_wmma_f32_16x16x32_bf16(
                false, a, false, bb, (short)0, acc[j], false, false);
        }
    }
    int mbase = mt * 16 + ((lane & 16) ? 8 : 0);
#pragma unroll
    for (int j = 0; j < NB; ++j) {
        int n = (ng * NB + j) * 16 + ncol;
        float bv = bias[n];
#pragma unroll
        for (int v = 0; v < 8; ++v)
            P[(size_t)(mbase + v) * 512 + n] = acc[j][v] + bv;
    }
}

// ---------------------------------------------------------------------------
// Data prep kernels
// ---------------------------------------------------------------------------
__global__ void k_f2bf(const float* __restrict__ s, bf16* __restrict__ d, int n) {
    int i = blockIdx.x * blockDim.x + threadIdx.x;
    if (i < n) d[i] = (bf16)s[i];
}

// out_w [111,512] -> padded [112,512] bf16 (row 111 zeroed)
__global__ void k_outw(const float* __restrict__ s, bf16* __restrict__ d) {
    int i = blockIdx.x * blockDim.x + threadIdx.x;
    if (i >= 112 * 512) return;
    int r = i >> 9;
    d[i] = (r < VOC_) ? (bf16)s[i] : (bf16)0.0f;
}

// feat_conv_w [O=512, I=512, 3,3] -> Wc [O, tap*512+ic] bf16
__global__ void k_convw(const float* __restrict__ s, bf16* __restrict__ d) {
    int i = blockIdx.x * blockDim.x + threadIdx.x;
    if (i >= 512 * KC_) return;
    int o = i / KC_, r = i % KC_;
    int tap = r >> 9, ic = r & 511;
    d[i] = (bf16)s[(size_t)o * KC_ + ic * 9 + tap];
}

// features NCHW f32 -> BHWC bf16
__global__ void k_fbhwc(const float* __restrict__ s, bf16* __restrict__ d) {
    int i = blockIdx.x * blockDim.x + threadIdx.x;
    if (i >= B_ * HW_ * FEAT_) return;
    int b = i >> 18;
    int rem = i & 262143;
    int y = rem >> 15;
    int rem2 = rem & 32767;
    int x = rem2 >> 9, c = rem2 & 511;
    d[i] = (bf16)s[((size_t)b * FEAT_ + c) * HW_ + y * 64 + x];
}

// teacher-forcing embedding gather -> Emb[480,512] bf16 (row = b*30+t)
__global__ void k_emb(const float* __restrict__ ew, const int* __restrict__ gt,
                      bf16* __restrict__ E) {
    int i = blockIdx.x * blockDim.x + threadIdx.x;
    if (i >= BT_ * RNN_) return;
    int bt = i >> 9, j = i & 511;
    int tok = gt[bt];                   // gt flat [B,T] == bt ordering
    E[i] = (bf16)ew[(size_t)tok * RNN_ + j];
}

// ---------------------------------------------------------------------------
// Fused LSTM-cell activation: gates = Gv[row] + (Gf ? Gf[b] : 0) + bih + bhh
// c = sig(f)*cprev + sig(i)*tanh(g); h = sig(o)*tanh(c)
// ---------------------------------------------------------------------------
__global__ void k_act(const float* __restrict__ Gv, const float* __restrict__ Gf,
                      const float* __restrict__ bih, const float* __restrict__ bhh,
                      const float* __restrict__ cprev, int cprev_per_b,
                      bf16* __restrict__ hbf, float* __restrict__ cout,
                      int rows, int tdiv) {
    int i = blockIdx.x * blockDim.x + threadIdx.x;
    if (i >= rows * 512) return;
    int r = i >> 9, j = i & 511;
    int b = r / tdiv;
    const float* g = Gv + (size_t)r * G_;
    const float* gfp = Gf ? (Gf + (size_t)b * G_) : nullptr;

    float gi = g[j]        + (gfp ? gfp[j]        : 0.f) + bih[j]        + bhh[j];
    float gf = g[j + 512]  + (gfp ? gfp[j + 512]  : 0.f) + bih[j + 512]  + bhh[j + 512];
    float gg = g[j + 1024] + (gfp ? gfp[j + 1024] : 0.f) + bih[j + 1024] + bhh[j + 1024];
    float go = g[j + 1536] + (gfp ? gfp[j + 1536] : 0.f) + bih[j + 1536] + bhh[j + 1536];

    float cp = cprev ? cprev[(size_t)(cprev_per_b ? b : r) * 512 + j] : 0.0f;
    float c = sigm(gf) * cp + sigm(gi) * ftanh(gg);
    float h = sigm(go) * ftanh(c);
    cout[i] = c;
    hbf[i]  = (bf16)h;
}

// ---------------------------------------------------------------------------
// Attention per (b,t): scores = sum_c tanh(P[b,hw,c]+S[bt,c])*pw[c];
// softmax over hw; glimpse[c] = sum_hw features[b,c,hw]*attn[hw]
// ---------------------------------------------------------------------------
__global__ void k_attn(const float* __restrict__ feat,   // [16,512,512] b,c,hw
                       const float* __restrict__ P,      // [8192,512] (b*hw, c)
                       const float* __restrict__ S,      // [480,512]
                       const float* __restrict__ pw,     // [512]
                       bf16* __restrict__ Gl) {          // [480,512] bf16
    __shared__ float sp[512];
    __shared__ float swp[512];
    __shared__ float sc[512];
    __shared__ float red[256];

    int bt = blockIdx.x;               // 0..479
    int b = bt / T_;
    int tid = threadIdx.x;             // 256 threads

    sp[tid]        = S[(size_t)bt * 512 + tid];
    sp[tid + 256]  = S[(size_t)bt * 512 + 256 + tid];
    swp[tid]       = pw[tid];
    swp[tid + 256] = pw[tid + 256];
    __syncthreads();

    // scores: each thread owns 2 spatial positions
#pragma unroll
    for (int half = 0; half < 2; ++half) {
        int hw = tid + half * 256;
        const float* fp = P + ((size_t)b * 512 + hw) * 512;
        float s = 0.0f;
        for (int c = 0; c < 512; c += 4) {
            float4 v = *(const float4*)(fp + c);
            s += ftanh(v.x + sp[c])     * swp[c]
               + ftanh(v.y + sp[c + 1]) * swp[c + 1]
               + ftanh(v.z + sp[c + 2]) * swp[c + 2]
               + ftanh(v.w + sp[c + 3]) * swp[c + 3];
        }
        sc[hw] = s;
    }
    __syncthreads();

    // softmax over 512 positions (stable)
    float m = fmaxf(sc[tid], sc[tid + 256]);
    red[tid] = m;
    __syncthreads();
    for (int s = 128; s > 0; s >>= 1) {
        if (tid < s) red[tid] = fmaxf(red[tid], red[tid + s]);
        __syncthreads();
    }
    float mx = red[0];
    __syncthreads();
    float e0 = __expf(sc[tid] - mx);
    float e1 = __expf(sc[tid + 256] - mx);
    sc[tid] = e0;
    sc[tid + 256] = e1;
    red[tid] = e0 + e1;
    __syncthreads();
    for (int s = 128; s > 0; s >>= 1) {
        if (tid < s) red[tid] += red[tid + s];
        __syncthreads();
    }
    float inv = 1.0f / red[0];
    __syncthreads();

    // glimpse: each thread owns 2 channels
#pragma unroll
    for (int half = 0; half < 2; ++half) {
        int c = tid + half * 256;
        const float* fp = feat + ((size_t)b * FEAT_ + c) * HW_;
        float g = 0.0f;
        for (int hw = 0; hw < 512; hw += 4) {
            float4 v = *(const float4*)(fp + hw);
            g += v.x * sc[hw] + v.y * sc[hw + 1] + v.z * sc[hw + 2] + v.w * sc[hw + 3];
        }
        Gl[(size_t)bt * 512 + c] = (bf16)(g * inv);
    }
}

// ---------------------------------------------------------------------------
// Host launch
// ---------------------------------------------------------------------------
static inline unsigned cdiv(unsigned a, unsigned b) { return (a + b - 1) / b; }
static inline unsigned gemm_blocks(int M, int N, int nbt) {
    int Mt = M >> 4, Nt = N >> 4;
    int Ng = Nt / nbt;
    return cdiv((unsigned)(Mt * Ng), 4);   // 4 waves per 128-thread block
}

extern "C" void kernel_launch(void* const* d_in, const int* in_sizes, int n_in,
                              void* d_out, int out_size, void* d_ws, size_t ws_size,
                              hipStream_t stream) {
    const float* features = (const float*)d_in[0];
    const float* encoded  = (const float*)d_in[1];
    const float* embed_w  = (const float*)d_in[2];
    const float* dec_Wih  = (const float*)d_in[3];
    // d_in[4] dec_Whh unused (multiplied by zero state)
    const float* dec_bih  = (const float*)d_in[5];
    const float* dec_bhh  = (const float*)d_in[6];
    const float* att_Wih  = (const float*)d_in[7];
    const float* att_Whh  = (const float*)d_in[8];
    const float* att_bih  = (const float*)d_in[9];
    const float* att_bhh  = (const float*)d_in[10];
    const float* fconv_w  = (const float*)d_in[11];
    const float* fconv_b  = (const float*)d_in[12];
    const float* state_w  = (const float*)d_in[13];
    const float* aproj_w  = (const float*)d_in[14];
    const float* out_w    = (const float*)d_in[15];
    const float* out_b    = (const float*)d_in[16];
    const int*   gt       = (const int*)d_in[17];
    float* out            = (float*)d_out;

    // workspace carve-up (256B aligned)
    char* w = (char*)d_ws;
    size_t off = 0;
    auto alloc = [&](size_t bytes) -> char* {
        char* p = w + off;
        off += (bytes + 255) & ~(size_t)255;
        return p;
    };
    bf16* enc_bf   = (bf16*)alloc(B_ * RNN_ * 2);
    bf16* decWih_b = (bf16*)alloc((size_t)G_ * RNN_ * 2);
    bf16* attWih_b = (bf16*)alloc((size_t)G_ * RNN_ * 2);
    bf16* attWhh_b = (bf16*)alloc((size_t)G_ * RNN_ * 2);
    bf16* state_b  = (bf16*)alloc((size_t)ATTN_ * RNN_ * 2);
    bf16* outw_b   = (bf16*)alloc((size_t)112 * RNN_ * 2);
    bf16* Wc_b     = (bf16*)alloc((size_t)512 * KC_ * 2);
    bf16* Fbhwc    = (bf16*)alloc((size_t)B_ * HW_ * FEAT_ * 2);
    float* Pfeat   = (float*)alloc((size_t)B_ * HW_ * ATTN_ * 4);
    float* Gdec    = (float*)alloc((size_t)B_ * G_ * 4);
    float* H0Whh   = (float*)alloc((size_t)B_ * G_ * 4);
    float* H0Wih   = (float*)alloc((size_t)B_ * G_ * 4);
    bf16* h0_bf    = (bf16*)alloc((size_t)B_ * RNN_ * 2);
    float* c0      = (float*)alloc((size_t)B_ * RNN_ * 4);
    bf16* Emb      = (bf16*)alloc((size_t)BT_ * RNN_ * 2);
    float* Gbig    = (float*)alloc((size_t)BT_ * G_ * 4);   // reused for gates1 & gates2
    bf16* h1_bf    = (bf16*)alloc((size_t)BT_ * RNN_ * 2);
    float* c1      = (float*)alloc((size_t)BT_ * RNN_ * 4);
    bf16* h2_bf    = (bf16*)alloc((size_t)BT_ * RNN_ * 2);
    float* Sproj   = (float*)alloc((size_t)BT_ * ATTN_ * 4);
    bf16* Gl       = (bf16*)alloc((size_t)BT_ * RNN_ * 2);
    (void)ws_size; (void)in_sizes; (void)n_in; (void)out_size;

    const unsigned TB = 256;
    // --- conversions ---
    k_f2bf<<<cdiv(B_ * RNN_, TB), TB, 0, stream>>>(encoded, enc_bf, B_ * RNN_);
    k_f2bf<<<cdiv(G_ * RNN_, TB), TB, 0, stream>>>(dec_Wih, decWih_b, G_ * RNN_);
    k_f2bf<<<cdiv(G_ * RNN_, TB), TB, 0, stream>>>(att_Wih, attWih_b, G_ * RNN_);
    k_f2bf<<<cdiv(G_ * RNN_, TB), TB, 0, stream>>>(att_Whh, attWhh_b, G_ * RNN_);
    k_f2bf<<<cdiv(ATTN_ * RNN_, TB), TB, 0, stream>>>(state_w, state_b, ATTN_ * RNN_);
    k_outw<<<cdiv(112 * 512, TB), TB, 0, stream>>>(out_w, outw_b);
    k_convw<<<cdiv(512 * KC_, TB), TB, 0, stream>>>(fconv_w, Wc_b);
    k_fbhwc<<<cdiv(B_ * HW_ * FEAT_, TB), TB, 0, stream>>>(features, Fbhwc);
    k_emb<<<cdiv(BT_ * RNN_, TB), TB, 0, stream>>>(embed_w, gt, Emb);

    // --- conv (feat_proj), hoisted out of the time loop ---
    k_conv<<<512, 256, 0, stream>>>(Fbhwc, Wc_b, fconv_b, Pfeat);  // 4096 waves

    // --- decoder LSTM once: h0, c0 ---
    k_gemm_t<NB><<<gemm_blocks(B_, G_, NB), 128, 0, stream>>>(
        enc_bf, decWih_b, Gdec, nullptr, B_, G_, RNN_, G_, G_);
    k_act<<<cdiv(B_ * 512, TB), TB, 0, stream>>>(Gdec, nullptr, dec_bih, dec_bhh,
                                                 nullptr, 0, h0_bf, c0, B_, 1);

    // --- step-invariant h0 projections ---
    k_gemm_t<NB><<<gemm_blocks(B_, G_, NB), 128, 0, stream>>>(
        h0_bf, attWhh_b, H0Whh, nullptr, B_, G_, RNN_, G_, G_);
    k_gemm_t<NB><<<gemm_blocks(B_, G_, NB), 128, 0, stream>>>(
        h0_bf, attWih_b, H0Wih, nullptr, B_, G_, RNN_, G_, G_);

    // --- all 30 time steps batched (no cross-step dependence) ---
    // gates1 = Emb @ Wih^T + h0@Whh^T + b   -> h1, c1
    k_gemm_t<NB><<<gemm_blocks(BT_, G_, NB), 128, 0, stream>>>(
        Emb, attWih_b, Gbig, nullptr, BT_, G_, RNN_, G_, G_);
    k_act<<<cdiv(BT_ * 512, TB), TB, 0, stream>>>(Gbig, H0Whh, att_bih, att_bhh,
                                                  c0, 1, h1_bf, c1, BT_, T_);
    // gates2 = h0 @ Wih^T + h1@Whh^T + b    -> h2 (cprev = c1 per row)
    k_gemm_t<NB><<<gemm_blocks(BT_, G_, NB), 128, 0, stream>>>(
        h1_bf, attWhh_b, Gbig, nullptr, BT_, G_, RNN_, G_, G_);
    k_act<<<cdiv(BT_ * 512, TB), TB, 0, stream>>>(Gbig, H0Wih, att_bih, att_bhh,
                                                  c1, 0, h2_bf, c1, BT_, T_);
    // state projection
    k_gemm_t<NB><<<gemm_blocks(BT_, ATTN_, NB), 128, 0, stream>>>(
        h2_bf, state_b, Sproj, nullptr, BT_, ATTN_, RNN_, ATTN_, ATTN_);
    // attention + glimpse
    k_attn<<<BT_, 256, 0, stream>>>(features, Pfeat, Sproj, aproj_w, Gl);
    // output head: [480,512] @ [111,512]^T + out_b -> d_out [16,30,111]
    // Nt=7 not divisible by 4 -> unblocked instantiation (tiny GEMM, 55 MFLOP)
    k_gemm_t<1><<<gemm_blocks(BT_, 112, 1), 128, 0, stream>>>(
        Gl, outw_b, out, out_b, BT_, 112, RNN_, VOC_, VOC_);
}
